// myNormedInvariantPolynomial_87531433493126
// MI455X (gfx1250) — matrix operand order repaired and checked by reading, not compiled, not run on hardware
//
#include <hip/hip_runtime.h>

typedef __attribute__((ext_vector_type(16))) _Float16 v16h;
typedef __attribute__((ext_vector_type(8)))  float    v8f;

#define LMAX 5
#define NSH 36
#define NB 20

__device__ inline float frcp(float x){ return __builtin_amdgcn_rcpf(x); }

// sqrt((2l+1)*(l-|m|)!/(l+|m|)!) * (sqrt2 if m!=0)
__constant__ float SH_C[6][6] = {
  {1.0f,0,0,0,0,0},
  {1.7320508f,1.7320508f,0,0,0,0},
  {2.2360680f,1.2909944f,0.6454972f,0,0,0},
  {2.6457513f,1.0801234f,0.3415650f,0.1394333f,0,0},
  {3.0f,0.9486833f,0.2236068f,0.0597614f,0.0211289f,0},
  {3.3166248f,0.8563488f,0.1618347f,0.0330344f,0.0077863f,0.0024623f}};

#define IDXP(l,m) ((l)*((l)+1)/2+(m))

__device__ inline void compute_sh(float vx,float vy,float vz,float r,float* sh){
  float inv = frcp(r);
  float xx=vx*inv, yy=vy*inv, zz=vz*inv;
  float phi = atan2f(yy,xx);
  float s2 = fminf(fmaxf(1.0f - zz*zz,0.0f),1.0f);
  float s = sqrtf(s2);
  float P[21];
  P[IDXP(0,0)] = 1.0f;
  #pragma unroll
  for(int m=1;m<=LMAX;m++) P[IDXP(m,m)] = (float)(2*m-1)*s*P[IDXP(m-1,m-1)];
  #pragma unroll
  for(int m=0;m<LMAX;m++) P[IDXP(m+1,m)] = (float)(2*m+1)*zz*P[IDXP(m,m)];
  #pragma unroll
  for(int l=2;l<=LMAX;l++){
    #pragma unroll
    for(int m=0;m<l-1;m++)
      P[IDXP(l,m)] = ((float)(2*l-1)*zz*P[IDXP(l-1,m)] - (float)(l-1+m)*P[IDXP(l-2,m)]) * (1.0f/(float)(l-m));
  }
  float cm[LMAX+1], sm[LMAX+1];
  cm[0]=1.0f; sm[0]=0.0f;
  float cp=cosf(phi), sp=sinf(phi);
  #pragma unroll
  for(int m=1;m<=LMAX;m++){ cm[m]=cm[m-1]*cp - sm[m-1]*sp; sm[m]=sm[m-1]*cp + cm[m-1]*sp; }
  int o=0;
  #pragma unroll
  for(int l=0;l<=LMAX;l++){
    #pragma unroll
    for(int m=-l;m<=l;m++){
      int am = m<0 ? -m : m;
      float ang = (m<0)? sm[am] : ((m>0)? cm[am] : 1.0f);
      sh[o++] = SH_C[l][am]*P[IDXP(l,am)]*ang;
    }
  }
}

// ---------------- device Wigner-3j generation (runs once per launch, 16 threads) ---------

__device__ double dfact_d(int n){ double r=1.0; for(int i=2;i<=n;i++) r*=(double)i; return r; }

__device__ double cg_d(int j1,int m1,int j2,int m2,int j3,int m3){
  if(m1+m2!=m3) return 0.0;
  double pref = sqrt((double)(2*j3+1)*dfact_d(j1+j2-j3)*dfact_d(j1-j2+j3)*dfact_d(-j1+j2+j3)/dfact_d(j1+j2+j3+1));
  pref *= sqrt(dfact_d(j1+m1)*dfact_d(j1-m1)*dfact_d(j2+m2)*dfact_d(j2-m2)*dfact_d(j3+m3)*dfact_d(j3-m3));
  double sacc=0.0;
  for(int k=0;k<=j1+j2-j3;k++){
    int d1=k, d2=j1+j2-j3-k, d3=j1-m1-k, d4=j2+m2-k, d5=j3-j2+m1+k, d6=j3-j1-m2+k;
    if(d1<0||d2<0||d3<0||d4<0||d5<0||d6<0) continue;
    double den = dfact_d(d1)*dfact_d(d2)*dfact_d(d3)*dfact_d(d4)*dfact_d(d5)*dfact_d(d6);
    sacc += ((k&1)? -1.0:1.0)/den;
  }
  return pref*sacc;
}

// nonzero entries of row i of u_real(l)  (<=2 per row)
__device__ int urow_d(int l,int i,int* cols,double* vr,double* vi){
  int m = i - l;
  const double rt = 0.7071067811865476;
  if(m==0){ cols[0]=l; vr[0]=1.0; vi[0]=0.0; return 1; }
  if(m>0){
    double sg = (m&1)? -1.0:1.0;
    cols[0]=l+m; vr[0]=sg*rt; vi[0]=0.0;
    cols[1]=l-m; vr[1]=rt;    vi[1]=0.0;
    return 2;
  }
  int am=-m;
  double sg = (am&1)? -1.0:1.0;
  cols[0]=l+am; vr[0]=0.0; vi[0]=-sg*rt;
  cols[1]=l-am; vr[1]=0.0; vi[1]=rt;
  return 2;
}

__device__ void w3j_dev(int j1,int j2,int j3,float* out){
  int n1=2*j1+1, n2=2*j2+1, n3=2*j3+1;
  int tot = n1*n2*n3;
  double Ar[512],Ai[512],Br[512],Bi[512];
  for(int i=0;i<tot;i++){ Ar[i]=0.0; Ai[i]=0.0; }
  for(int m1=-j1;m1<=j1;m1++)
    for(int m2=-j2;m2<=j2;m2++){
      int m3 = -(m1+m2);
      if(m3< -j3 || m3>j3) continue;
      int e = j1-j2-m3;
      double sg = (e&1)? -1.0:1.0;
      Ar[((j1+m1)*n2+(j2+m2))*n3+(j3+m3)] = sg/sqrt((double)(2*j3+1))*cg_d(j1,m1,j2,m2,j3,-m3);
    }
  int cols[2]; double vr[2],vi[2];
  // dim0 transform by conj(U(j1))
  for(int i=0;i<tot;i++){ Br[i]=0.0; Bi[i]=0.0; }
  for(int i=0;i<n1;i++){
    int nz = urow_d(j1,i,cols,vr,vi);
    for(int q=0;q<nz;q++){
      double ur=vr[q], ui=-vi[q];
      int a=cols[q];
      for(int bc=0;bc<n2*n3;bc++){
        double ar=Ar[a*n2*n3+bc], ai=Ai[a*n2*n3+bc];
        Br[i*n2*n3+bc] += ur*ar - ui*ai;
        Bi[i*n2*n3+bc] += ur*ai + ui*ar;
      }
    }
  }
  // dim1 transform by conj(U(j2))
  for(int i=0;i<tot;i++){ Ar[i]=0.0; Ai[i]=0.0; }
  for(int i=0;i<n1;i++)
    for(int j=0;j<n2;j++){
      int nz = urow_d(j2,j,cols,vr,vi);
      for(int q=0;q<nz;q++){
        double ur=vr[q], ui=-vi[q];
        int b=cols[q];
        for(int c=0;c<n3;c++){
          double br=Br[(i*n2+b)*n3+c], bi=Bi[(i*n2+b)*n3+c];
          Ar[(i*n2+j)*n3+c] += ur*br - ui*bi;
          Ai[(i*n2+j)*n3+c] += ur*bi + ui*br;
        }
      }
    }
  // dim2 transform by conj(U(j3)), keep real part
  for(int ij=0;ij<n1*n2;ij++)
    for(int k=0;k<n3;k++){
      int nz = urow_d(j3,k,cols,vr,vi);
      double accr=0.0;
      for(int q=0;q<nz;q++){
        double ur=vr[q], ui=-vi[q];
        int c=cols[q];
        accr += ur*Ar[ij*n3+c] - ui*Ai[ij*n3+c];
      }
      out[ij*n3+k] = (float)accr;
    }
}

__constant__ int L1A[11]={0,1,1,2,2,2,3,3,3,4,4};
__constant__ int L2A[11]={2,1,3,0,2,4,1,3,5,2,4};
__constant__ int OFF2A[11]={0,25,70,175,200,325,550,655,900,1285,1510};
__constant__ int OFF1A[5]={0,1,10,35,84};

__global__ void init_w3j_k(float* w3j1, float* w3j2){
  int t = threadIdx.x;
  if(t<5){
    w3j_dev(0, t, t, w3j1 + OFF1A[t]);      // shape (1,2l+1,2l+1), stored [j][m]
  } else if(t<16){
    int p = t-5;
    w3j_dev(L1A[p], L2A[p], 2, w3j2 + OFF2A[p]);
  }
}

// ---------------- weight fragment pre-swizzle (B-matrix WMMA layout) ----------------
// B frag (32x16, f16): lane L holds column n=L%16, K = 8*(L/16)+j for j<8, 16+8*(L/16)+(j-8) for j>=8.
__global__ void init_wfrag_k(const float* __restrict__ w1, const float* __restrict__ w2,
                             _Float16* __restrict__ w1f, _Float16* __restrict__ w2f){
  int t = blockIdx.x*blockDim.x + threadIdx.x;
  if(t < 4*32*16){
    int j = t & 15; int lane = (t>>4)&31; int n = t>>9;
    int k = ((lane<16)?0:8) + (j&7) + ((j&8)?16:0);
    int col = n*16 + (lane&15);
    float v = (k<20 && col<50) ? w1[k*50+col] : 0.0f;
    w1f[t] = (_Float16)v;
  }
  if(t < 2*32*16){
    int j = t & 15; int lane = (t>>4)&31; int kt = t>>9;
    int k = kt*32 + ((lane<16)?0:8) + (j&7) + ((j&8)?16:0);
    int col = lane&15;
    float v = (k<50 && col<5) ? w2[k*5+col] : 0.0f;
    w2f[t] = (_Float16)v;
  }
}

__global__ void zero_k(unsigned int* p, unsigned int n){
  unsigned int i = blockIdx.x*blockDim.x + threadIdx.x;
  if(i<n) p[i]=0u;
}

// ---------------- pass 1: SH + radial MLP (WMMA) + tp1 + scatter ----------------
__global__ __launch_bounds__(256) void edge_pass1(
    const float* __restrict__ x, const float* __restrict__ pos,
    const int* __restrict__ ei,
    const _Float16* __restrict__ w1f, const _Float16* __restrict__ w2f,
    const float* __restrict__ w3j1,
    float* __restrict__ node, unsigned int* __restrict__ counts, int E)
{
  __shared__ _Float16 embT[8][32][32];   // per-wave 32-edge x 32(K-pad) emb tile
  __shared__ _Float16 hT[8][32][64];     // per-wave hidden tile (N-pad 64)
  __shared__ float    twT[8][32][16];    // per-wave tp weights tile
  __shared__ float    w3j1s[165];
  for(int i=threadIdx.x;i<165;i+=256) w3j1s[i]=w3j1[i];

  int wave = threadIdx.x>>5, lane = threadIdx.x&31;
  int half = lane>>4, lcol = lane&15;
  int e = (blockIdx.x*8 + wave)*32 + lane;
  bool valid = e < E;
  int ec = valid ? e : E-1;
  int from = ei[2*ec], to = ei[2*ec+1];
  float vx = pos[3*to+0]-pos[3*from+0];
  float vy = pos[3*to+1]-pos[3*from+1];
  float vz = pos[3*to+2]-pos[3*from+2];
  float r = sqrtf(vx*vx+vy*vy+vz*vz);
  float sh[NSH];
  compute_sh(vx,vy,vz,r,sh);

  // soft one-hot radial embedding -> LDS (f16, zero-padded K 20->32)
  const float invstep = 21.0f/3.0f;
  const float step = 3.0f/21.0f;
  #pragma unroll
  for(int i=0;i<32;i++){
    float val = 0.0f;
    if(i<NB){
      float diff = (r - (float)(i+1)*step)*invstep;
      float a = diff+1.0f, b = 1.0f-diff;
      float sa = a>0.0f ? __expf(-frcp(a)) : 0.0f;
      float sb = b>0.0f ? __expf(-frcp(b)) : 0.0f;
      val = 8.4335731f*sa*sb;   // 1.14136*e^2 * sus(diff+1)*sus(1-diff)
    }
    embT[wave][lane][i] = (_Float16)val;
  }
  __syncthreads();

  // MLP1: (32x32) @ (32x64)  via 2x4 tiles of v_wmma_f32_16x16x32_f16
  v8f zero = {};
  v8f acc[2][4];
  #pragma unroll
  for(int m=0;m<2;m++){
    #pragma unroll
    for(int n=0;n<4;n++) acc[m][n]=zero;
  }
  #pragma unroll
  for(int m=0;m<2;m++){
    v16h a;
    {
      const _Float16* p = &embT[wave][m*16 + lcol][half*8];
      #pragma unroll
      for(int j=0;j<8;j++){ a[j]=p[j]; a[j+8]=p[j+16]; }
    }
    #pragma unroll
    for(int n=0;n<4;n++){
      v16h b = *(const v16h*)(w1f + ((n*32+lane)<<4));
      acc[m][n] = __builtin_amdgcn_wmma_f32_16x16x32_f16(false,a,false,b,(short)0,acc[m][n],false,false);
    }
  }
  // h = silu(acc/sqrt(20)) * SILU_NORM  -> f16 LDS tile  (fast sigmoid via v_rcp_f32)
  #pragma unroll
  for(int m=0;m<2;m++){
    #pragma unroll
    for(int n=0;n<4;n++){
      #pragma unroll
      for(int q=0;q<8;q++){
        float t = acc[m][n][q]*0.2236068f;
        float hv = t*1.679f*frcp(1.0f+__expf(-t));
        hT[wave][m*16 + half*8 + q][n*16 + lcol] = (_Float16)hv;
      }
    }
  }
  __syncthreads();

  // MLP2: (32x64) @ (64x16) via 2 M-tiles x 2 K-steps
  v8f a2[2]; a2[0]=zero; a2[1]=zero;
  #pragma unroll
  for(int m=0;m<2;m++){
    #pragma unroll
    for(int kt=0;kt<2;kt++){
      v16h a;
      const _Float16* p = &hT[wave][m*16+lcol][kt*32 + half*8];
      #pragma unroll
      for(int j=0;j<8;j++){ a[j]=p[j]; a[j+8]=p[j+16]; }
      v16h b = *(const v16h*)(w2f + ((kt*32+lane)<<4));
      a2[m] = __builtin_amdgcn_wmma_f32_16x16x32_f16(false,a,false,b,(short)0,a2[m],false,false);
    }
  }
  #pragma unroll
  for(int m=0;m<2;m++){
    #pragma unroll
    for(int q=0;q<8;q++)
      twT[wave][m*16+half*8+q][lcol] = a2[m][q]*0.14142136f;  // /sqrt(50)
  }
  __syncthreads();

  // tp1: feat_l = sqrt(2l+1)*tw_l*x_to*(sh_l @ W3j(0,l,l)); scatter-add /sqrt(6)
  if(valid){
    float xv = x[to];
    float tw[5];
    #pragma unroll
    for(int l=0;l<5;l++) tw[l]=twT[wave][lane][l];
    const float AL[5]={1.0f,1.7320508f,2.2360680f,2.6457513f,3.0f};
    const int   O1[5]={0,1,10,35,84};
    float base = xv*0.40824829f;
    #pragma unroll
    for(int l=0;l<5;l++){
      const int n=2*l+1;
      float wl = AL[l]*tw[l]*base;
      const float* W = w3j1s + O1[l];
      #pragma unroll
      for(int mm=0;mm<n;mm++){
        float accd=0.0f;
        #pragma unroll
        for(int j=0;j<n;j++) accd += sh[l*l+j]*W[j*n+mm];
        atomicAdd(&node[from*25 + l*l + mm], wl*accd);
      }
    }
    atomicAdd(&counts[to], 1u);
  }
}

// ---------------- pass 2: tp2 + scatter ----------------
__global__ __launch_bounds__(256) void edge_pass2(
    const float* __restrict__ pos, const int* __restrict__ ei,
    const float* __restrict__ node, const float* __restrict__ w3j2,
    const float* __restrict__ tp2w, float* __restrict__ node2, int E)
{
  __shared__ float Ws[1915];
  __shared__ float wp[11];
  for(int i=threadIdx.x;i<1915;i+=256) Ws[i]=w3j2[i];
  if(threadIdx.x<11) wp[threadIdx.x]=tp2w[threadIdx.x]*0.67419986f; // *sqrt(5/11)
  __syncthreads();
  int e = blockIdx.x*256 + threadIdx.x;
  if(e>=E) return;
  int from = ei[2*e], to = ei[2*e+1];
  float vx = pos[3*to+0]-pos[3*from+0];
  float vy = pos[3*to+1]-pos[3*from+1];
  float vz = pos[3*to+2]-pos[3*from+2];
  float r = sqrtf(vx*vx+vy*vy+vz*vz);
  float sh[NSH];
  compute_sh(vx,vy,vz,r,sh);
  float f[25];
  #pragma unroll
  for(int i=0;i<25;i++) f[i]=node[to*25+i];   // node array is L2-resident (10MB)
  float o0=0,o1=0,o2=0,o3=0,o4=0;
  const int l1a[11]={0,1,1,2,2,2,3,3,3,4,4};
  const int l2a[11]={2,1,3,0,2,4,1,3,5,2,4};
  const int offa[11]={0,25,70,175,200,325,550,655,900,1285,1510};
  #pragma unroll
  for(int p=0;p<11;p++){
    const int l1=l1a[p], l2=l2a[p];
    const int n2=2*l2+1;
    float w = wp[p];
    const float* W = Ws + offa[p];
    #pragma unroll
    for(int i=0;i<2*l1+1;i++){
      float fi = f[l1*l1+i]*w;
      #pragma unroll
      for(int j=0;j<n2;j++){
        float t = fi*sh[l2*l2+j];
        const float* Wij = W + (i*n2+j)*5;
        o0 += t*Wij[0]; o1 += t*Wij[1]; o2 += t*Wij[2]; o3 += t*Wij[3]; o4 += t*Wij[4];
      }
    }
  }
  atomicAdd(&node2[from*5+0],o0);
  atomicAdd(&node2[from*5+1],o1);
  atomicAdd(&node2[from*5+2],o2);
  atomicAdd(&node2[from*5+3],o3);
  atomicAdd(&node2[from*5+4],o4);
}

// ---------------- argmax(bincount) + output ----------------
__global__ void argmax_k(const unsigned int* __restrict__ counts, int N, int* co){
  __shared__ unsigned int bc[256];
  __shared__ int bi[256];
  unsigned int best=0; int besti=0x7fffffff;
  for(int i=threadIdx.x;i<N;i+=256){
    unsigned int c=counts[i];
    if(c>best || (c==best && i<besti)){ best=c; besti=i; }
  }
  bc[threadIdx.x]=best; bi[threadIdx.x]=besti;
  __syncthreads();
  for(int s=128;s>0;s>>=1){
    if(threadIdx.x<(unsigned)s){
      if(bc[threadIdx.x+s]>bc[threadIdx.x] ||
         (bc[threadIdx.x+s]==bc[threadIdx.x] && bi[threadIdx.x+s]<bi[threadIdx.x])){
        bc[threadIdx.x]=bc[threadIdx.x+s]; bi[threadIdx.x]=bi[threadIdx.x+s];
      }
    }
    __syncthreads();
  }
  if(threadIdx.x==0) *co = bi[0];
}

__global__ void writeout_k(const float* __restrict__ node2, const int* __restrict__ co,
                           float* __restrict__ out){
  if(threadIdx.x<5) out[threadIdx.x]=node2[(*co)*5+threadIdx.x];
}

// ---------------- host launcher ----------------
extern "C" void kernel_launch(void* const* d_in, const int* in_sizes, int n_in,
                              void* d_out, int out_size, void* d_ws, size_t ws_size,
                              hipStream_t stream) {
  const float* x    = (const float*)d_in[0];
  const float* pos  = (const float*)d_in[1];
  const int*   ei   = (const int*)d_in[2];
  const float* w1   = (const float*)d_in[3];
  const float* w2   = (const float*)d_in[4];
  const float* tp2w = (const float*)d_in[5];
  const int N = in_sizes[1]/3;
  const int E = in_sizes[2]/2;

  size_t off = 0;
  auto take = [&](size_t bytes)->size_t{
    off = (off + 255) & ~(size_t)255;
    size_t r = off; off += bytes; return r;
  };
  size_t o_node   = take((size_t)N*25*4);
  size_t o_node2  = take((size_t)N*5*4);
  size_t o_counts = take((size_t)N*4);
  size_t o_w3j1   = take(165*4);
  size_t o_w3j2   = take(1915*4);
  size_t o_w1f    = take(4*32*16*2);
  size_t o_w2f    = take(2*32*16*2);
  size_t o_co     = take(4);
  (void)ws_size; (void)n_in; (void)out_size;

  char* ws = (char*)d_ws;
  float* nodep   = (float*)(ws + o_node);
  float* node2p  = (float*)(ws + o_node2);
  unsigned int* countsp = (unsigned int*)(ws + o_counts);
  float* w3j1p   = (float*)(ws + o_w3j1);
  float* w3j2p   = (float*)(ws + o_w3j2);
  _Float16* w1fp = (_Float16*)(ws + o_w1f);
  _Float16* w2fp = (_Float16*)(ws + o_w2f);
  int* cop       = (int*)(ws + o_co);

  // zero node accumulators + counts (single span from ws start through counts)
  unsigned int zwords = (unsigned int)((o_counts + (size_t)N*4)/4);
  zero_k<<<(zwords+255)/256, 256, 0, stream>>>((unsigned int*)ws, zwords);

  init_w3j_k<<<1, 16, 0, stream>>>(w3j1p, w3j2p);
  init_wfrag_k<<<8, 256, 0, stream>>>(w1, w2, w1fp, w2fp);

  int blocks = (E + 255)/256;
  edge_pass1<<<blocks, 256, 0, stream>>>(x, pos, ei, w1fp, w2fp, w3j1p, nodep, countsp, E);
  edge_pass2<<<blocks, 256, 0, stream>>>(pos, ei, nodep, w3j2p, tp2w, node2p, E);
  argmax_k<<<1, 256, 0, stream>>>(countsp, N, cop);
  writeout_k<<<1, 32, 0, stream>>>(node2p, cop, (float*)d_out);
}
